// DeepSpeedTransformerLayer_23613730193412
// MI455X (gfx1250) — matrix operand order repaired and checked by reading, not compiled, not run on hardware
//
#include <hip/hip_runtime.h>

// ---------------------------------------------------------------------------
// Types for CDNA5 WMMA (wave32)
// ---------------------------------------------------------------------------
typedef __attribute__((ext_vector_type(16))) __bf16 v16bf;
typedef __attribute__((ext_vector_type(8)))  float  v8f;
typedef __attribute__((ext_vector_type(4)))  int    v4i;

// LDS tile row stride in bf16 elements. 72*2 = 144 B per row: 16 B aligned so
// ds_load_b128 works from any row start, and 36 words/row spreads rows 0..15
// across distinct LDS banks (36*r mod 64 all distinct) -> conflict-free frags.
static constexpr int LDT = 72;
static constexpr int SCL = 516;   // score LDS stride (floats), 16B aligned

// ---------------------------------------------------------------------------
// Async global->LDS copies (gfx1250 GLOBAL_LOAD_ASYNC_TO_LDS_B128, ASYNCcnt).
// Builtin signature (from hipcc diagnostic): (v4i addrspace(1)*, v4i addrspace(3)*, imm, imm)
// Guarded: falls back to load+ds_store if the builtins are not declared.
// ---------------------------------------------------------------------------
#if defined(__has_builtin)
#  if __has_builtin(__builtin_amdgcn_global_load_async_to_lds_b128) && \
      __has_builtin(__builtin_amdgcn_s_wait_asynccnt)
#    define USE_ASYNC_LDS 1
#  endif
#endif
#ifndef USE_ASYNC_LDS
#  define USE_ASYNC_LDS 0
#endif

#if USE_ASYNC_LDS
typedef __attribute__((address_space(1))) v4i gv4i;
typedef __attribute__((address_space(3))) v4i lv4i;
#endif

__device__ __forceinline__ void async_copy16(const __bf16* g, __bf16* l) {
#if USE_ASYNC_LDS
  __builtin_amdgcn_global_load_async_to_lds_b128((gv4i*)g, (lv4i*)l, 0, 0);
#else
  *(uint4*)l = *(const uint4*)g;
#endif
}

__device__ __forceinline__ void async_wait_all() {
#if USE_ASYNC_LDS
  __builtin_amdgcn_s_wait_asynccnt(0);
#endif
}

__device__ __forceinline__ v8f wmma_bf16(v16bf a, v16bf b, v8f c) {
  return __builtin_amdgcn_wmma_f32_16x16x32_bf16(false, a, false, b, (short)0, c,
                                                 false, false);
}

// A-fragment (16x32 bf16, M x K): lane L (m=L&15, hi=L>>4) holds row m,
// K = [kk+hi*8, +8) then [kk+16+hi*8, +8)  -> two contiguous 16B LDS loads.
__device__ __forceinline__ v16bf frag_a(const __bf16* base, int row0, int lane, int kk) {
  int m = lane & 15, hi = lane >> 4;
  const __bf16* p = base + (row0 + m) * LDT;
  v16bf f;
  *((uint4*)&f)       = *(const uint4*)(p + kk + hi * 8);
  *(((uint4*)&f) + 1) = *(const uint4*)(p + kk + 16 + hi * 8);
  return f;
}

// B-fragment (32x16 bf16, K x N), stored row-per-N in LDS (i.e. base[n][k]):
// lane L (n=L&15, hi=L>>4) holds column n, K = [kk+hi*8, +8) then [kk+16+hi*8, +8).
__device__ __forceinline__ v16bf frag_b(const __bf16* base, int n0, int lane, int kk) {
  int n = lane & 15, hi = lane >> 4;
  const __bf16* p = base + (n0 + n) * LDT + kk + hi * 8;
  v16bf f;
  *((uint4*)&f)       = *(const uint4*)(p);
  *(((uint4*)&f) + 1) = *(const uint4*)(p + 16);
  return f;
}

// A-fragment built from an f32 LDS strip (probabilities), converted to bf16.
__device__ __forceinline__ v16bf frag_a_f32(const float* base, int row0, int lane, int kbase) {
  int m = lane & 15, hi = lane >> 4;
  const float* p = base + (row0 + m) * SCL + kbase + hi * 8;
  v16bf f;
#pragma unroll
  for (int j = 0; j < 8; ++j) f[j] = (__bf16)p[j];
#pragma unroll
  for (int j = 0; j < 8; ++j) f[8 + j] = (__bf16)p[16 + j];
  return f;
}

// ---------------------------------------------------------------------------
// fp32 -> bf16 convert
// ---------------------------------------------------------------------------
__global__ __launch_bounds__(256) void cvt_bf16(const float* __restrict__ in,
                                                __bf16* __restrict__ out, int n) {
  int i = blockIdx.x * blockDim.x + threadIdx.x;
  int stride = gridDim.x * blockDim.x;
  int nv = n >> 2;
  const float4* in4 = (const float4*)in;
  for (int v = i; v < nv; v += stride) {
    float4 f = in4[v];
    out[v * 4 + 0] = (__bf16)f.x;
    out[v * 4 + 1] = (__bf16)f.y;
    out[v * 4 + 2] = (__bf16)f.z;
    out[v * 4 + 3] = (__bf16)f.w;
  }
}

// ---------------------------------------------------------------------------
// GEMM: C[M,N] = A[M,K] @ W[N,K]^T + bias, optional tanh-GELU.
// Block tile 128x128, BK=64, 8 waves, wave tile 32x64 (2x4 WMMA tiles).
// Double-buffered LDS, async global->LDS prefetch of the next K-slice.
// ---------------------------------------------------------------------------
template <bool GELU>
__global__ __launch_bounds__(256) void gemm_bf16(
    const __bf16* __restrict__ A, const __bf16* __restrict__ W,
    const float* __restrict__ bias, float* __restrict__ outF,
    __bf16* __restrict__ outB, int M, int N, int K) {
  __shared__ __bf16 As[2][128 * LDT];
  __shared__ __bf16 Ws[2][128 * LDT];
  const int t = threadIdx.x;
  const int lane = t & 31, w = t >> 5;
  const int wm = w >> 1, wn = w & 1;
  const int mBase = blockIdx.y * 128, nBase = blockIdx.x * 128;

  v8f acc[2][4] = {};

  auto issue_tiles = [&](int buf, int k0) {
#pragma unroll
    for (int c = 0; c < 4; ++c) {   // 128x64 bf16 per tile, 4 x 16B per thread
      int idx = (t + c * 256) * 8;
      int r = idx >> 6, col = idx & 63;
      async_copy16(A + (size_t)(mBase + r) * K + k0 + col, &As[buf][r * LDT + col]);
      async_copy16(W + (size_t)(nBase + r) * K + k0 + col, &Ws[buf][r * LDT + col]);
    }
  };

  issue_tiles(0, 0);
  async_wait_all();
  __syncthreads();

  int cur = 0;
  for (int k0 = 0; k0 < K; k0 += 64) {
    if (k0 + 64 < K) issue_tiles(cur ^ 1, k0 + 64);  // prefetch next slice
#pragma unroll
    for (int kk = 0; kk < 64; kk += 32) {
      v16bf a0 = frag_a(As[cur], wm * 32, lane, kk);
      v16bf a1 = frag_a(As[cur], wm * 32 + 16, lane, kk);
#pragma unroll
      for (int j = 0; j < 4; ++j) {
        v16bf bj = frag_b(Ws[cur], wn * 64 + j * 16, lane, kk);
        acc[0][j] = wmma_bf16(a0, bj, acc[0][j]);
        acc[1][j] = wmma_bf16(a1, bj, acc[1][j]);
      }
    }
    async_wait_all();
    __syncthreads();
    cur ^= 1;
  }

  const int lm = lane & 15, hi = lane >> 4;
#pragma unroll
  for (int i = 0; i < 2; ++i)
#pragma unroll
    for (int j = 0; j < 4; ++j) {
      int col = nBase + wn * 64 + j * 16 + lm;
      float bv = bias[col];
#pragma unroll
      for (int r = 0; r < 8; ++r) {
        int row = mBase + wm * 32 + i * 16 + r + hi * 8;
        float v = acc[i][j][r] + bv;
        if (GELU) {
          float u = 0.7978845608028654f * (v + 0.044715f * v * v * v);
          v = 0.5f * v * (1.0f + tanhf(u));
        }
        size_t o = (size_t)row * N + col;
        if (outF) outF[o] = v;
        if (outB) outB[o] = (__bf16)v;
      }
    }
}

// ---------------------------------------------------------------------------
// Attention: per block = (qb 64-row strip, head h, batch b).
// scores strip 64x512 kept in LDS f32; exact softmax; P@V via WMMA.
// qkv layout: [B*S, 3H] bf16 (q at col 0, k at 1024, v at 2048).
// ---------------------------------------------------------------------------
__global__ __launch_bounds__(256) void attn_kernel(const __bf16* __restrict__ qkv,
                                                   const float* __restrict__ mask,
                                                   __bf16* __restrict__ ctx) {
  extern __shared__ char smem[];
  __bf16* Qs    = (__bf16*)(smem);               //  9216 B
  __bf16* KV0   = (__bf16*)(smem + 9216);        //  9216 B
  __bf16* KV1   = (__bf16*)(smem + 18432);       //  9216 B
  float*  Sc    = (float*)(smem + 27648);        // 64*516*4 = 132096 B
  float*  maskS = (float*)(smem + 159744);       //  2048 B
  float*  redA  = (float*)(smem + 161792);       //  1024 B
  float*  redB  = (float*)(smem + 162816);       //  1024 B
  float*  rowinv= (float*)(smem + 163840);       //   256 B  (total 164096)
  __bf16* KV[2] = {KV0, KV1};

  const int t = threadIdx.x;
  const int lane = t & 31, w = t >> 5;
  const int qb = blockIdx.x, h = blockIdx.y, b = blockIdx.z;
  const size_t rowQ0 = (size_t)b * 512 + qb * 64;
  const size_t rowK0 = (size_t)b * 512;

  // mask row (additive, [B,1,1,S])
  maskS[t]       = mask[b * 512 + t];
  maskS[t + 256] = mask[b * 512 + t + 256];

  auto issue_k = [&](int buf, int kt) {          // K tile 64x64 -> LDS
#pragma unroll
    for (int c = 0; c < 2; ++c) {
      int idx = (t + c * 256) * 8;
      int r = idx >> 6, col = idx & 63;
      async_copy16(qkv + (rowK0 + kt * 64 + r) * 3072 + 1024 + h * 64 + col,
                   &KV[buf][r * LDT + col]);
    }
  };

  // Q strip -> LDS (64x64 bf16) + first K tile, async
#pragma unroll
  for (int c = 0; c < 2; ++c) {
    int idx = (t + c * 256) * 8;
    int r = idx >> 6, col = idx & 63;
    async_copy16(qkv + (rowQ0 + r) * 3072 + h * 64 + col, &Qs[r * LDT + col]);
  }
  issue_k(0, 0);
  async_wait_all();
  __syncthreads();

  // ---- phase 1: scores = Q @ K^T (double-buffered K tiles) ----------------
  const int tid0 = w * 2;
  const int qt = tid0 >> 2;          // tiles tid0, tid0+1 share qt
  const int nt0 = tid0 & 3;          // and have adjacent nt/dt
  const int lm = lane & 15, hi = lane >> 4;

  for (int kt = 0; kt < 8; ++kt) {
    int cur = kt & 1;
    if (kt + 1 < 8) issue_k(cur ^ 1, kt + 1);
    v8f s0 = {}, s1 = {};
#pragma unroll
    for (int kk = 0; kk < 64; kk += 32) {
      v16bf a  = frag_a(Qs, qt * 16, lane, kk);
      v16bf b0 = frag_b(KV[cur], nt0 * 16, lane, kk);
      v16bf b1 = frag_b(KV[cur], (nt0 + 1) * 16, lane, kk);
      s0 = wmma_bf16(a, b0, s0);
      s1 = wmma_bf16(a, b1, s1);
    }
    int col0 = kt * 64 + nt0 * 16 + lm;
#pragma unroll
    for (int r = 0; r < 8; ++r) {
      int rr = qt * 16 + r + hi * 8;
      Sc[rr * SCL + col0]      = s0[r];
      Sc[rr * SCL + col0 + 16] = s1[r];
    }
    async_wait_all();
    __syncthreads();
  }

  // ---- phase 2: softmax (scale 1/8, +mask); keep unnormalized, 1/sum later
  {
    const int r = t >> 2, g = t & 3;
    const float scale = 0.125f;
    float m = -3.0e30f;
    for (int c = g * 128; c < g * 128 + 128; ++c) {
      float v = Sc[r * SCL + c] * scale + maskS[c];
      Sc[r * SCL + c] = v;
      m = fmaxf(m, v);
    }
    redA[r * 4 + g] = m;
    __syncthreads();
    m = fmaxf(fmaxf(redA[r * 4 + 0], redA[r * 4 + 1]),
              fmaxf(redA[r * 4 + 2], redA[r * 4 + 3]));
    float s = 0.0f;
    for (int c = g * 128; c < g * 128 + 128; ++c) {
      float e = __expf(Sc[r * SCL + c] - m);
      Sc[r * SCL + c] = e;
      s += e;
    }
    redB[r * 4 + g] = s;
    __syncthreads();
    s = redB[r * 4 + 0] + redB[r * 4 + 1] + redB[r * 4 + 2] + redB[r * 4 + 3];
    if (g == 0) rowinv[r] = 1.0f / s;
  }

  // ---- phase 3: ctx = P @ V ----------------------------------------------
  v8f c0 = {}, c1 = {};
  for (int kt = 0; kt < 8; ++kt) {
    __syncthreads();
#pragma unroll
    for (int c = 0; c < 2; ++c) {    // V tile loaded TRANSPOSED: KV0[d][s_k]
      int idx = (t + c * 256) * 8;
      int r = idx >> 6, col = idx & 63;
      uint4 raw = *(const uint4*)(qkv + (rowK0 + kt * 64 + r) * 3072 + 2048 + h * 64 + col);
      const __bf16* e = (const __bf16*)&raw;
#pragma unroll
      for (int j = 0; j < 8; ++j) KV0[(col + j) * LDT + r] = e[j];
    }
    __syncthreads();
#pragma unroll
    for (int kk = 0; kk < 64; kk += 32) {
      v16bf a  = frag_a_f32(Sc, qt * 16, lane, kt * 64 + kk);
      v16bf b0 = frag_b(KV0, nt0 * 16, lane, kk);
      v16bf b1 = frag_b(KV0, (nt0 + 1) * 16, lane, kk);
      c0 = wmma_bf16(a, b0, c0);
      c1 = wmma_bf16(a, b1, c1);
    }
  }

#pragma unroll
  for (int r = 0; r < 8; ++r) {
    int rr = qt * 16 + r + hi * 8;
    float inv = rowinv[rr];
    size_t o = (rowQ0 + rr) * 1024 + h * 64 + nt0 * 16 + lm;
    ctx[o]      = (__bf16)(c0[r] * inv);
    ctx[o + 16] = (__bf16)(c1[r] * inv);
  }
}

// ---------------------------------------------------------------------------
// Residual + LayerNorm: y = LN(a+b)*w + bias, f32 out and optional bf16 out.
// One row (H=1024) per block, 256 threads x 4 elems.
// ---------------------------------------------------------------------------
__global__ __launch_bounds__(256) void ln_kernel(
    const float* __restrict__ a, const float* __restrict__ res,
    const float* __restrict__ w, const float* __restrict__ bias,
    float* __restrict__ outF, __bf16* __restrict__ outB) {
  __shared__ float red[8];
  __shared__ float red2[8];
  const int row = blockIdx.x, t = threadIdx.x;
  const float* pa = a + (size_t)row * 1024;
  const float* pb = res + (size_t)row * 1024;

  float v[4];
  float s = 0.0f;
#pragma unroll
  for (int i = 0; i < 4; ++i) {
    v[i] = pa[t + i * 256] + pb[t + i * 256];
    s += v[i];
  }
  for (int off = 16; off > 0; off >>= 1) s += __shfl_down(s, off);
  if ((t & 31) == 0) red[t >> 5] = s;
  __syncthreads();
  float mu = 0.0f;
#pragma unroll
  for (int i = 0; i < 8; ++i) mu += red[i];
  mu *= (1.0f / 1024.0f);

  float q = 0.0f;
#pragma unroll
  for (int i = 0; i < 4; ++i) {
    float d = v[i] - mu;
    q += d * d;
  }
  for (int off = 16; off > 0; off >>= 1) q += __shfl_down(q, off);
  if ((t & 31) == 0) red2[t >> 5] = q;
  __syncthreads();
  float var = 0.0f;
#pragma unroll
  for (int i = 0; i < 8; ++i) var += red2[i];
  var *= (1.0f / 1024.0f);
  float inv = rsqrtf(var + 1e-12f);

#pragma unroll
  for (int i = 0; i < 4; ++i) {
    int c = t + i * 256;
    float y = (v[i] - mu) * inv * w[c] + bias[c];
    size_t o = (size_t)row * 1024 + c;
    if (outF) outF[o] = y;
    if (outB) outB[o] = (__bf16)y;
  }
}

// ---------------------------------------------------------------------------
// Host launcher
// ---------------------------------------------------------------------------
extern "C" void kernel_launch(void* const* d_in, const int* in_sizes, int n_in,
                              void* d_out, int out_size, void* d_ws, size_t ws_size,
                              hipStream_t stream) {
  (void)in_sizes; (void)n_in; (void)out_size; (void)ws_size;
  const float* x    = (const float*)d_in[0];
  const float* mask = (const float*)d_in[1];
  const float* qkvw = (const float*)d_in[2];
  const float* qkvb = (const float*)d_in[3];
  const float* ow   = (const float*)d_in[4];
  const float* ob   = (const float*)d_in[5];
  const float* anw  = (const float*)d_in[6];
  const float* anb  = (const float*)d_in[7];
  const float* iw   = (const float*)d_in[8];
  const float* ib   = (const float*)d_in[9];
  const float* pw   = (const float*)d_in[10];
  const float* pb   = (const float*)d_in[11];
  const float* nw   = (const float*)d_in[12];
  const float* nb   = (const float*)d_in[13];
  float* out = (float*)d_out;

  // workspace layout (all sizes 256B-aligned by construction)
  char* ws = (char*)d_ws;
  size_t off = 0;
  auto take = [&](size_t bytes) { char* p = ws + off; off += bytes; return p; };
  __bf16* x_b    = (__bf16*)take((size_t)8192 * 1024 * 2);       // 16 MB
  __bf16* qkvw_b = (__bf16*)take((size_t)3072 * 1024 * 2);       //  6 MB
  __bf16* ow_b   = (__bf16*)take((size_t)1024 * 1024 * 2);       //  2 MB
  __bf16* iw_b   = (__bf16*)take((size_t)4096 * 1024 * 2);       //  8 MB
  __bf16* pw_b   = (__bf16*)take((size_t)1024 * 4096 * 2);       //  8 MB
  char*   big    = take((size_t)8192 * 3072 * 2 + (size_t)8192 * 1024 * 2); // 64 MB
  __bf16* qkv_b  = (__bf16*)big;                                  // [8192,3072]
  __bf16* ctx_b  = (__bf16*)(big + (size_t)8192 * 3072 * 2);      // [8192,1024]
  __bf16* h1_b   = (__bf16*)big;           // reuses qkv+ctx after attention
  float*  att_f  = (float*)take((size_t)8192 * 1024 * 4);        // 32 MB (also ff2)
  float*  ln1_f  = (float*)take((size_t)8192 * 1024 * 4);        // 32 MB
  __bf16* ln1_b  = (__bf16*)take((size_t)8192 * 1024 * 2);       // 16 MB

  // fp32 -> bf16 (x + weights)
  cvt_bf16<<<2048, 256, 0, stream>>>(x,    x_b,    8192 * 1024);
  cvt_bf16<<<2048, 256, 0, stream>>>(qkvw, qkvw_b, 3072 * 1024);
  cvt_bf16<<<1024, 256, 0, stream>>>(ow,   ow_b,   1024 * 1024);
  cvt_bf16<<<2048, 256, 0, stream>>>(iw,   iw_b,   4096 * 1024);
  cvt_bf16<<<2048, 256, 0, stream>>>(pw,   pw_b,   1024 * 4096);

  // QKV projection: [8192,1024] x [3072,1024]^T -> bf16 qkv
  gemm_bf16<false><<<dim3(24, 64), 256, 0, stream>>>(
      x_b, qkvw_b, qkvb, nullptr, qkv_b, 8192, 3072, 1024);

  // attention (softmax strip in LDS): 8 q-strips x 16 heads x 16 batches
  attn_kernel<<<dim3(8, 16, 16), 256, 164096, stream>>>(qkv_b, mask, ctx_b);

  // output projection -> f32 attn_out
  gemm_bf16<false><<<dim3(8, 64), 256, 0, stream>>>(
      ctx_b, ow_b, ob, att_f, nullptr, 8192, 1024, 1024);

  // LN(x + attn_out) -> ln1 (f32 + bf16)
  ln_kernel<<<8192, 256, 0, stream>>>(x, att_f, anw, anb, ln1_f, ln1_b);

  // FF1 + GELU -> bf16 h1 (aliases qkv/ctx region)
  gemm_bf16<true><<<dim3(32, 64), 256, 0, stream>>>(
      ln1_b, iw_b, ib, nullptr, h1_b, 8192, 4096, 1024);

  // FF2 -> f32 ff2 (reuses att_f)
  gemm_bf16<false><<<dim3(8, 64), 256, 0, stream>>>(
      h1_b, pw_b, pb, att_f, nullptr, 8192, 1024, 4096);

  // final LN(ff2 + ln1) -> d_out
  ln_kernel<<<8192, 256, 0, stream>>>(att_f, ln1_f, nw, nb, out, nullptr);
}
